// CrossAttention_6743098655540
// MI455X (gfx1250) — matrix operand order repaired and checked by reading, not compiled, not run on hardware
//
#include <hip/hip_runtime.h>
#include <hip/hip_bf16.h>

// ---------------------------------------------------------------------------
// CDNA5 (gfx1250) two-head flash cross-attention, all GEMMs via
// v_wmma_f32_16x16x32_f16 (wave32). BN1/BN2/BN-last and the kq interleave are
// folded into projection weights/biases; sigmoid gate fused into the output
// projection epilogue.  K/V tiles in the attention kernel are staged with
// double-buffered async-to-LDS copies (ASYNCcnt pipelining).
// ---------------------------------------------------------------------------

typedef __attribute__((ext_vector_type(16))) _Float16 v16h;
typedef __attribute__((ext_vector_type(8)))  _Float16 v8h;
typedef __attribute__((ext_vector_type(8)))  float    v8f;

#define Bsz   4
#define Ntok  4096          // 64*64
#define Cin   256
#define Dh    64            // head dim
#define NH    2             // heads

__device__ __forceinline__ v8f vzero8() {
  v8f z;
#pragma unroll
  for (int i = 0; i < 8; ++i) z[i] = 0.0f;
  return z;
}

// A operand (16x32 f16): lane = row M (lane&15); lanes0-15 hold K 0-7 & 16-23,
// lanes16-31 hold K 8-15 & 24-31 (two 16B chunks).
__device__ __forceinline__ v16h load_a(const _Float16* base, int stride,
                                       int row_base, int kbase, int lane) {
  int row = row_base + (lane & 15);
  int k0  = kbase + ((lane >> 4) << 3);
  const _Float16* p = base + row * stride + k0;
  v8h lo = *(const v8h*)(p);
  v8h hi = *(const v8h*)(p + 16);
  v16h r;
#pragma unroll
  for (int i = 0; i < 8; ++i) { r[i] = lo[i]; r[i + 8] = hi[i]; }
  return r;
}

// B operand (32x16 f16): lane = col N (lane&15); lanes0-15 hold K 0-7 & 16-23,
// lanes16-31 hold K 8-15 & 24-31 (two 16B chunks). `base` is laid out [N][K].
__device__ __forceinline__ v16h load_b(const _Float16* base, int stride,
                                       int col_base, int kbase, int lane) {
  int col = col_base + (lane & 15);
  int k0  = kbase + ((lane >> 4) << 3);
  const _Float16* p = base + col * stride + k0;
  v8h lo = *(const v8h*)(p);
  v8h hi = *(const v8h*)(p + 16);
  v16h r;
#pragma unroll
  for (int i = 0; i < 8; ++i) { r[i] = lo[i]; r[i + 8] = hi[i]; }
  return r;
}

__device__ __forceinline__ v8f wmma16(v16h a, v16h b, v8f c) {
  return __builtin_amdgcn_wmma_f32_16x16x32_f16(false, a, false, b,
                                                (short)0, c, false, false);
}

// Async global->LDS copy, 16B per lane (GLOBAL_LOAD_ASYNC_TO_LDS_B128, GV mode).
// lds_off = LDS byte offset (low 32 bits of generic __shared__ pointer),
// gaddr   = 64-bit global address in a VGPR pair.  Tracked by ASYNCcnt.
__device__ __forceinline__ void async_copy_b128(unsigned lds_off, const void* gaddr) {
  asm volatile("global_load_async_to_lds_b128 %0, %1, off"
               :
               : "v"(lds_off), "v"(gaddr)
               : "memory");
}

// ---------------------------------------------------------------------------
// Kernel 0: fold BNs into projection weights; build f16 weights + f32 biases.
// ---------------------------------------------------------------------------
__global__ void k_prep(const float* __restrict__ kq1_w, const float* __restrict__ kq2_w,
                       const float* __restrict__ v_w,   const float* __restrict__ out_w,
                       const float* __restrict__ bn1_w, const float* __restrict__ bn1_b,
                       const float* __restrict__ bn1_m, const float* __restrict__ bn1_v,
                       const float* __restrict__ bn2_w, const float* __restrict__ bn2_b,
                       const float* __restrict__ bn2_m, const float* __restrict__ bn2_v,
                       const float* __restrict__ bnl_w, const float* __restrict__ bnl_b,
                       const float* __restrict__ bnl_m, const float* __restrict__ bnl_v,
                       const float* __restrict__ kq1_b, const float* __restrict__ kq2_b,
                       const float* __restrict__ v_b,   const float* __restrict__ out_b,
                       _Float16* __restrict__ Wa, _Float16* __restrict__ Wb,
                       _Float16* __restrict__ Wv, _Float16* __restrict__ Wo,
                       float* __restrict__ bA, float* __restrict__ bB,
                       float* __restrict__ bV, float* __restrict__ bO) {
  const float EPSf = 1e-5f;
  int tid   = blockIdx.x * blockDim.x + threadIdx.x;
  int total = gridDim.x * blockDim.x;

  for (int i = tid; i < 128 * Cin; i += total) {        // Wa/Wb/Wv: [128][256]
    int c = i & (Cin - 1);
    float i1 = bn1_w[c] * rsqrtf(bn1_v[c] + EPSf);
    float i2 = bn2_w[c] * rsqrtf(bn2_v[c] + EPSf);
    Wa[i] = (_Float16)(kq1_w[i] * i1);
    Wb[i] = (_Float16)(kq2_w[i] * i2);
    Wv[i] = (_Float16)(v_w[i]);
  }
  for (int i = tid; i < Cin * 128; i += total) {        // Wo: [256][128]
    int c = i >> 7;
    float iL = bnl_w[c] * rsqrtf(bnl_v[c] + EPSf);
    Wo[i] = (_Float16)(out_w[i] * iL);
  }
  for (int j = tid; j < 128; j += total) {              // folded biases
    float s1 = kq1_b[j], s2 = kq2_b[j];
    for (int c = 0; c < Cin; ++c) {
      float i1 = bn1_w[c] * rsqrtf(bn1_v[c] + EPSf);
      float i2 = bn2_w[c] * rsqrtf(bn2_v[c] + EPSf);
      s1 += (bn1_b[c] - bn1_m[c] * i1) * kq1_w[j * Cin + c];
      s2 += (bn2_b[c] - bn2_m[c] * i2) * kq2_w[j * Cin + c];
    }
    bA[j] = s1; bB[j] = s2; bV[j] = v_b[j];
  }
  for (int c = tid; c < Cin; c += total) {
    float iL = bnl_w[c] * rsqrtf(bnl_v[c] + EPSf);
    bO[c] = out_b[c] * iL + (bnl_b[c] - bnl_m[c] * iL);
  }
}

// ---------------------------------------------------------------------------
// Kernel 1: NCHW f32 -> [B][N][C] f16, LDS tile transpose (coalesced both ways)
// grid (N/64, C/64, B), block 256
// ---------------------------------------------------------------------------
__global__ void k_nchw_to_nhc(const float* __restrict__ src, _Float16* __restrict__ dst) {
  __shared__ float tile[64][65];
  int b  = blockIdx.z;
  int n0 = blockIdx.x * 64;
  int c0 = blockIdx.y * 64;
  int tn  = threadIdx.x & 63;
  int tc4 = threadIdx.x >> 6;
#pragma unroll
  for (int i = 0; i < 16; ++i) {
    int c = tc4 * 16 + i;
    tile[c][tn] = src[((size_t)b * Cin + (c0 + c)) * Ntok + n0 + tn];
  }
  __syncthreads();
  int wc  = threadIdx.x & 63;
  int wn4 = threadIdx.x >> 6;
#pragma unroll
  for (int i = 0; i < 16; ++i) {
    int n = wn4 * 16 + i;
    dst[((size_t)b * Ntok + (n0 + n)) * Cin + c0 + wc] = (_Float16)tile[wc][n];
  }
}

// ---------------------------------------------------------------------------
// Kernel 2: projection GEMM  C[64n x 128j] = X[64 x 256] * W[128 x 256]^T + b
// X tile staged in LDS; W streamed from L2 (64KB, shared by all 256 blocks).
// mode 0: rgb->kq1 (even interleave)  mode 1: dsm->kq2 (odd)  mode 2: fus->v
// grid (N/64, B), block 128 (4 waves).  64 WMMA / wave.
// ---------------------------------------------------------------------------
__global__ void k_proj(const _Float16* __restrict__ X,   // [B][N][256] f16
                       const _Float16* __restrict__ W,   // [128][256] f16
                       const float*    __restrict__ bias,// [128]
                       _Float16* __restrict__ dstK,      // K base (or V base)
                       _Float16* __restrict__ dstQ,      // Q base (unused mode 2)
                       int mode) {
  __shared__ __align__(16) _Float16 Xs[64 * Cin];        // 32 KB
  int b  = blockIdx.y;
  int n0 = blockIdx.x * 64;
  int tid  = threadIdx.x;
  int lane = tid & 31;
  int wave = tid >> 5;

  const uint4* xg = (const uint4*)(X + ((size_t)b * Ntok + n0) * Cin);
  uint4* xs = (uint4*)Xs;
  for (int i = tid; i < 64 * Cin / 8; i += 128) xs[i] = xg[i];
  __syncthreads();

  v8f acc[8];
#pragma unroll
  for (int k = 0; k < 8; ++k) acc[k] = vzero8();

#pragma unroll
  for (int ks = 0; ks < 8; ++ks) {
    v16h a = load_a(Xs, Cin, wave * 16, ks * 32, lane);
#pragma unroll
    for (int blk = 0; blk < 8; ++blk) {
      v16h bm = load_b(W, Cin, blk * 16, ks * 32, lane);   // global (L2-hot)
      acc[blk] = wmma16(a, bm, acc[blk]);
    }
  }

  int rbase = (lane >> 4) << 3;
  int cl    = lane & 15;
#pragma unroll
  for (int blk = 0; blk < 8; ++blk) {
    int j = blk * 16 + cl;
    float bj = bias[j];
#pragma unroll
    for (int r = 0; r < 8; ++r) {
      int n = n0 + wave * 16 + rbase + r;
      _Float16 val = (_Float16)(acc[blk][r] + bj);
      if (mode < 2) {
        int cch  = 2 * j + mode;        // interleave fold
        int g    = cch >> 6;            // 0:k1 1:q1 2:k2 3:q2
        int d    = cch & 63;
        int head = g >> 1;
        _Float16* dst = (g & 1) ? dstQ : dstK;
        dst[(((size_t)head * Bsz + b) * Ntok + n) * Dh + d] = val;
      } else {
        int head = j >> 6;
        int d    = j & 63;
        dstK[(((size_t)head * Bsz + b) * Ntok + n) * Dh + d] = val;
      }
    }
  }
}

// ---------------------------------------------------------------------------
// Kernel 3: flash attention.  block = 128 thr (4 waves), each wave owns 16
// query rows.  K/V tiles double-buffered via global_load_async_to_lds_b128:
// tile t+1 streams into LDS (ASYNCcnt) while tile t runs 16 WMMA + softmax.
// grid (N/64, heads, B)
// ---------------------------------------------------------------------------
__global__ void k_attn(const _Float16* __restrict__ Q,   // [2][B][N][64]
                       const _Float16* __restrict__ K,
                       const _Float16* __restrict__ V,
                       _Float16* __restrict__ O) {       // [B][N][128]
  __shared__ __align__(16) _Float16 Qs[64 * Dh];         // 8 KB
  __shared__ __align__(16) _Float16 Ks2[2][64 * Dh];     // 16 KB (double buffer)
  __shared__ __align__(16) _Float16 Vr2[2][64 * Dh];     // 16 KB (double buffer, raw)
  __shared__ __align__(16) _Float16 Vs[Dh * 64];         // 8 KB  (transposed [d][key])
  __shared__ __align__(16) _Float16 Ps[4][16 * 64];      // 8 KB  (per-wave P)

  int n0   = blockIdx.x * 64;
  int head = blockIdx.y;
  int b    = blockIdx.z;
  int tid  = threadIdx.x;
  int lane = tid & 31;
  int wave = tid >> 5;
  size_t base = ((size_t)head * Bsz + b) * Ntok * Dh;
  const int NT = Ntok / 64;                              // 64 key tiles

  // LDS byte offsets for async VDST operands (flat->LDS aperture: low 32 bits)
  unsigned ksOff[2], vrOff[2];
  ksOff[0] = (unsigned)(size_t)(&Ks2[0][0]);
  ksOff[1] = (unsigned)(size_t)(&Ks2[1][0]);
  vrOff[0] = (unsigned)(size_t)(&Vr2[0][0]);
  vrOff[1] = (unsigned)(size_t)(&Vr2[1][0]);

  // Q tile -> LDS (regular loads, once)
  const uint4* qg = (const uint4*)(Q + base + (size_t)n0 * Dh);
  uint4* qs = (uint4*)Qs;
  for (int i = tid; i < 64 * Dh / 8; i += 128) qs[i] = qg[i];

  // prologue: async-stage tile 0 into buffer 0 (8 x b128 per wave)
  {
    const char* kg = (const char*)(K + base);
    const char* vg = (const char*)(V + base);
#pragma unroll
    for (int c = 0; c < 4; ++c) {
      int byteoff = (c * 128 + tid) * 16;
      async_copy_b128(ksOff[0] + byteoff, kg + byteoff);
      async_copy_b128(vrOff[0] + byteoff, vg + byteoff);
    }
  }
  __syncthreads();                                       // Qs visible

  v16h qa0 = load_a(Qs, Dh, wave * 16, 0, lane);
  v16h qa1 = load_a(Qs, Dh, wave * 16, 32, lane);

  float m8[8], l8[8];
#pragma unroll
  for (int r = 0; r < 8; ++r) { m8[r] = -1e30f; l8[r] = 0.0f; }
  v8f oacc[4];
#pragma unroll
  for (int k = 0; k < 4; ++k) oacc[k] = vzero8();

  const float scale = 0.125f;                            // 1/sqrt(64)
  int rbase = (lane >> 4) << 3;
  int cl    = lane & 15;

  for (int t = 0; t < NT; ++t) {
    int cur = t & 1;
    if (t + 1 < NT) {                                    // stream tile t+1
      const char* kg = (const char*)(K + base + (size_t)(t + 1) * 64 * Dh);
      const char* vg = (const char*)(V + base + (size_t)(t + 1) * 64 * Dh);
#pragma unroll
      for (int c = 0; c < 4; ++c) {
        int byteoff = (c * 128 + tid) * 16;
        async_copy_b128(ksOff[1 - cur] + byteoff, kg + byteoff);
        async_copy_b128(vrOff[1 - cur] + byteoff, vg + byteoff);
      }
      // in-order completion: <=8 outstanding  =>  tile t fully in LDS
      asm volatile("s_wait_asynccnt 0x8" ::: "memory");
    } else {
      asm volatile("s_wait_asynccnt 0x0" ::: "memory");
    }
    __syncthreads();      // all waves: tile t resident; prev Vs/Ps reads retired

    // transpose V tile for B-operand layout ([d][key])
    const _Float16* vr = &Vr2[cur][0];
    for (int i = tid; i < 64 * Dh; i += 128) {
      int key = i >> 6, d = i & 63;
      Vs[d * 64 + key] = vr[i];
    }

    // ---- S = Q K^T (per wave: 16 x 64) ----
    const _Float16* kt = &Ks2[cur][0];
    v8f s[4];
#pragma unroll
    for (int blk = 0; blk < 4; ++blk) {
      v8f a = vzero8();
      a = wmma16(qa0, load_b(kt, Dh, blk * 16, 0, lane), a);
      a = wmma16(qa1, load_b(kt, Dh, blk * 16, 32, lane), a);
      s[blk] = a;
    }
    __syncthreads();                                     // Vs ready block-wide

    // ---- online softmax (rows live across 16-lane halves) ----
    float mnew[8];
#pragma unroll
    for (int r = 0; r < 8; ++r) mnew[r] = m8[r];
#pragma unroll
    for (int blk = 0; blk < 4; ++blk)
#pragma unroll
      for (int r = 0; r < 8; ++r) {
        s[blk][r] *= scale;
        mnew[r] = fmaxf(mnew[r], s[blk][r]);
      }
#pragma unroll
    for (int r = 0; r < 8; ++r)
#pragma unroll
      for (int mask = 1; mask < 16; mask <<= 1)
        mnew[r] = fmaxf(mnew[r], __shfl_xor(mnew[r], mask, 32));

    float alpha[8], ls[8];
#pragma unroll
    for (int r = 0; r < 8; ++r) {
      alpha[r] = __expf(m8[r] - mnew[r]);
      m8[r] = mnew[r];
      ls[r] = 0.0f;
    }
#pragma unroll
    for (int blk = 0; blk < 4; ++blk)
#pragma unroll
      for (int r = 0; r < 8; ++r) {
        float p = __expf(s[blk][r] - m8[r]);
        s[blk][r] = p;
        ls[r] += p;
      }
#pragma unroll
    for (int r = 0; r < 8; ++r) {
#pragma unroll
      for (int mask = 1; mask < 16; mask <<= 1)
        ls[r] += __shfl_xor(ls[r], mask, 32);
      l8[r] = l8[r] * alpha[r] + ls[r];
    }
#pragma unroll
    for (int blk = 0; blk < 4; ++blk)
#pragma unroll
      for (int r = 0; r < 8; ++r) oacc[blk][r] *= alpha[r];

    // ---- P -> LDS (C-layout -> A-layout round trip), then O += P V ----
    _Float16* Pw = &Ps[wave][0];
#pragma unroll
    for (int blk = 0; blk < 4; ++blk)
#pragma unroll
      for (int r = 0; r < 8; ++r)
        Pw[(rbase + r) * 64 + blk * 16 + cl] = (_Float16)s[blk][r];

    v16h pa0 = load_a(Pw, 64, 0, 0, lane);          // same-wave RAW: hw waitcnt
    v16h pa1 = load_a(Pw, 64, 0, 32, lane);
#pragma unroll
    for (int blk = 0; blk < 4; ++blk) {
      oacc[blk] = wmma16(pa0, load_b(Vs, 64, blk * 16, 0, lane), oacc[blk]);
      oacc[blk] = wmma16(pa1, load_b(Vs, 64, blk * 16, 32, lane), oacc[blk]);
    }
  }

  // ---- normalize, store O[b][n][head*64+d] ----
#pragma unroll
  for (int blk = 0; blk < 4; ++blk)
#pragma unroll
    for (int r = 0; r < 8; ++r) {
      int n = n0 + wave * 16 + rbase + r;
      int d = blk * 16 + cl;
      float v = oacc[blk][r] / l8[r];
      O[((size_t)b * Ntok + n) * 128 + head * Dh + d] = (_Float16)v;
    }
}

// ---------------------------------------------------------------------------
// Kernel 4: output projection + BN-last (folded) + sigmoid gate.
// C[64n x 256c] = O2[64 x 128] * Wo[256 x 128]^T + bO ; out = sig(C)*fus*g + fus
// grid (N/64, B), block 128 (4 waves).  64 WMMA / wave.
// ---------------------------------------------------------------------------
__global__ void k_outproj(const _Float16* __restrict__ O2,  // [B][N][128] f16
                          const _Float16* __restrict__ Wo,  // [256][128] f16
                          const float* __restrict__ bO,     // [256]
                          const float* __restrict__ fusion, // [B][256][4096]
                          const float* __restrict__ gamma,  // [1]
                          float* __restrict__ out) {        // [B][256][4096]
  __shared__ __align__(16) _Float16 Os[64 * 128];           // 16 KB
  __shared__ __align__(16) _Float16 Cs[64 * 264];           // 33 KB (padded)
  int b  = blockIdx.y;
  int n0 = blockIdx.x * 64;
  int tid  = threadIdx.x;
  int lane = tid & 31;
  int wave = tid >> 5;

  const uint4* og = (const uint4*)(O2 + ((size_t)b * Ntok + n0) * 128);
  uint4* osp = (uint4*)Os;
  for (int i = tid; i < 64 * 128 / 8; i += 128) osp[i] = og[i];
  __syncthreads();

  v8f acc[16];
#pragma unroll
  for (int k = 0; k < 16; ++k) acc[k] = vzero8();

#pragma unroll
  for (int ks = 0; ks < 4; ++ks) {
    v16h a = load_a(Os, 128, wave * 16, ks * 32, lane);
#pragma unroll
    for (int blk = 0; blk < 16; ++blk) {
      v16h bm = load_b(Wo, 128, blk * 16, ks * 32, lane);   // global (L2-hot)
      acc[blk] = wmma16(a, bm, acc[blk]);
    }
  }

  int rbase = (lane >> 4) << 3;
  int cl    = lane & 15;
#pragma unroll
  for (int blk = 0; blk < 16; ++blk) {
    int c = blk * 16 + cl;
    float bb = bO[c];
#pragma unroll
    for (int r = 0; r < 8; ++r)
      Cs[(wave * 16 + rbase + r) * 264 + c] = (_Float16)(acc[blk][r] + bb);
  }
  __syncthreads();

  float g0 = gamma[0];
  int nl = tid & 63;                     // coalesced over n in NCHW
  for (int cc = (tid >> 6); cc < Cin; cc += 2) {
    float val = (float)Cs[nl * 264 + cc];
    float sg = 1.0f / (1.0f + __expf(-val));
    size_t gi = ((size_t)b * Cin + cc) * Ntok + n0 + nl;
    float f = fusion[gi];
    out[gi] = sg * f * g0 + f;
  }
}

// ---------------------------------------------------------------------------
extern "C" void kernel_launch(void* const* d_in, const int* in_sizes, int n_in,
                              void* d_out, int out_size, void* d_ws, size_t ws_size,
                              hipStream_t stream) {
  (void)in_sizes; (void)n_in; (void)out_size; (void)ws_size;

  const float* fusion = (const float*)d_in[0];
  const float* rgb    = (const float*)d_in[1];
  const float* dsm    = (const float*)d_in[2];
  const float* bn1_w = (const float*)d_in[3],  *bn1_b = (const float*)d_in[4];
  const float* bn1_m = (const float*)d_in[5],  *bn1_v = (const float*)d_in[6];
  const float* bn2_w = (const float*)d_in[7],  *bn2_b = (const float*)d_in[8];
  const float* bn2_m = (const float*)d_in[9],  *bn2_v = (const float*)d_in[10];
  const float* kq1_w = (const float*)d_in[11], *kq1_b = (const float*)d_in[12];
  const float* kq2_w = (const float*)d_in[13], *kq2_b = (const float*)d_in[14];
  const float* v_w   = (const float*)d_in[15], *v_b   = (const float*)d_in[16];
  const float* out_w = (const float*)d_in[17], *out_b = (const float*)d_in[18];
  const float* bnl_w = (const float*)d_in[19], *bnl_b = (const float*)d_in[20];
  const float* bnl_m = (const float*)d_in[21], *bnl_v = (const float*)d_in[22];
  const float* gamma = (const float*)d_in[23];
  float* outp = (float*)d_out;

  // workspace carve-out (256B aligned)
  size_t off = 0;
  auto carve = [&](size_t bytes) -> void* {
    off = (off + 255) & ~(size_t)255;
    void* p = (char*)d_ws + off;
    off += bytes;
    return p;
  };
  const size_t xbytes = (size_t)Bsz * Ntok * Cin * sizeof(_Float16);    // 8 MB
  const size_t hbytes = (size_t)NH * Bsz * Ntok * Dh * sizeof(_Float16);// 4 MB
  _Float16* Xrgb = (_Float16*)carve(xbytes);
  _Float16* Xdsm = (_Float16*)carve(xbytes);
  _Float16* Xfus = (_Float16*)carve(xbytes);
  _Float16* Wa = (_Float16*)carve(128 * Cin * 2);
  _Float16* Wb = (_Float16*)carve(128 * Cin * 2);
  _Float16* Wv = (_Float16*)carve(128 * Cin * 2);
  _Float16* Wo = (_Float16*)carve(Cin * 128 * 2);
  float* bA = (float*)carve(128 * 4);
  float* bB = (float*)carve(128 * 4);
  float* bV = (float*)carve(128 * 4);
  float* bO = (float*)carve(Cin * 4);
  _Float16* Qb = (_Float16*)carve(hbytes);
  _Float16* Kb = (_Float16*)carve(hbytes);
  _Float16* Vb = (_Float16*)carve(hbytes);
  _Float16* O2 = (_Float16*)carve((size_t)Bsz * Ntok * 128 * 2);        // 4 MB

  k_prep<<<dim3(64), dim3(256), 0, stream>>>(
      kq1_w, kq2_w, v_w, out_w,
      bn1_w, bn1_b, bn1_m, bn1_v, bn2_w, bn2_b, bn2_m, bn2_v,
      bnl_w, bnl_b, bnl_m, bnl_v,
      kq1_b, kq2_b, v_b, out_b,
      Wa, Wb, Wv, Wo, bA, bB, bV, bO);

  dim3 tg(Ntok / 64, Cin / 64, Bsz);
  k_nchw_to_nhc<<<tg, 256, 0, stream>>>(rgb,    Xrgb);
  k_nchw_to_nhc<<<tg, 256, 0, stream>>>(dsm,    Xdsm);
  k_nchw_to_nhc<<<tg, 256, 0, stream>>>(fusion, Xfus);

  dim3 pg(Ntok / 64, Bsz);
  k_proj<<<pg, 128, 0, stream>>>(Xrgb, Wa, bA, Kb, Qb, 0);
  k_proj<<<pg, 128, 0, stream>>>(Xdsm, Wb, bB, Kb, Qb, 1);
  k_proj<<<pg, 128, 0, stream>>>(Xfus, Wv, bV, Vb, Vb, 2);

  k_attn<<<dim3(Ntok / 64, NH, Bsz), 128, 0, stream>>>(Qb, Kb, Vb, O2);

  k_outproj<<<pg, 128, 0, stream>>>(O2, Wo, bO, fusion, gamma, outp);
}
